// MlpWithAttention_41412074668087
// MI455X (gfx1250) — matrix operand order repaired and checked by reading, not compiled, not run on hardware
//
#include <hip/hip_runtime.h>

typedef __attribute__((ext_vector_type(16))) _Float16 v16h;
typedef __attribute__((ext_vector_type(8)))  float    v8f;

#define LDSTRIDE  36         // 16B-aligned rows; half-waves still on disjoint banks (288%64==32)
#define LN_EPS    1e-5f
#define NEG_SLOPE 0.01f
#define NFRAG     14         // preconverted f16 B-fragments per block

// ---- WMMA wrapper: D = A(16x32 f16) * B(32x16 f16) + C(16x16 f32) ----
__device__ __forceinline__ v8f wmma_f16(v16h a, v16h b, v8f c) {
  return __builtin_amdgcn_wmma_f32_16x16x32_f16(
      /*neg_a=*/false, a, /*neg_b=*/false, b,
      /*c_mod=*/(short)0, c, /*reuse_a=*/false, /*reuse_b=*/false);
}

// ---- wave-local LDS ordering: DS ops are in-order per wave (ISA 7.3), so a
// compiler-level fence is enough for a wave-private tile; emits no barrier HW ops.
__device__ __forceinline__ void wave_sync() {
  __builtin_amdgcn_fence(__ATOMIC_ACQ_REL, "wavefront");
  __builtin_amdgcn_wave_barrier();
}

// ---- A fragment (16x32, f16) from row-major f32 matrix P (ld floats/row) ----
// ISA 7.12.2: lanes 0-15 -> M=lane, K = {kb..kb+7, kb+16..kb+23}, kb=0
//             lanes 16-31 -> M=lane-16, same with kb=8
__device__ __forceinline__ v16h load_afrag(const float* __restrict__ P, int ld,
                                           int kbase, int lane) {
  const int m  = lane & 15;
  const int kb = kbase + ((lane >> 4) << 3);
  const float* p = P + m * ld + kb;
  v16h a;
#pragma unroll
  for (int e = 0; e < 8; ++e) {
    a[e]     = (_Float16)p[e];
    a[8 + e] = (_Float16)p[16 + e];
  }
  return a;
}

// ---- B fragment: two ds_load_b128 from the preconverted LDS weight pool ----
__device__ __forceinline__ v16h bfrag(const _Float16* WF, int frag, int lane) {
  return *(const v16h*)(WF + frag * 512 + lane * 16);
}

// ---- C init: broadcast bias[n] into all 8 accumulator rows ----
__device__ __forceinline__ v8f bias_c(const float* __restrict__ bias, int nbase, int lane) {
  const float v = bias[nbase + (lane & 15)];
  v8f c;
#pragma unroll
  for (int r = 0; r < 8; ++r) c[r] = v;
  return c;
}

// ---- C init for attention: c[r] + gamma*bv[n]  (residual + scaled bias) ----
__device__ __forceinline__ v8f resid_c(v8f c, float gamma,
                                       const float* __restrict__ bv, int nbase, int lane) {
  const float b = bv[nbase + (lane & 15)];
  v8f o;
#pragma unroll
  for (int r = 0; r < 8; ++r) o[r] = fmaf(gamma, b, c[r]);
  return o;
}

// ---- store C frag (16x16 f32) to LDS tile T[16][LDSTRIDE] at column block nbase ----
// lanes 0-15: M = r ; lanes 16-31: M = 8+r ; N = nbase + (lane&15)
__device__ __forceinline__ void store_c(float* T, int nbase, int lane, v8f c) {
  const int col = nbase + (lane & 15);
  const int rb  = (lane >> 4) << 3;
#pragma unroll
  for (int r = 0; r < 8; ++r) T[(rb + r) * LDSTRIDE + col] = c[r];
}

// ---- LayerNorm(32) + LeakyReLU, in-place on LDS tile; full wave active ----
// lane pair (l, l+16) shares row l&15: each handles 16 features, shfl_xor combines.
__device__ __forceinline__ void ln_leaky(float* T, const float* gLds,
                                         const float* bLds, int lane) {
  const int row = lane & 15;
  const int jb  = (lane >> 4) << 4;          // 0 or 16
  float* p = T + row * LDSTRIDE + jb;
  float v[16];
  float s = 0.f, s2 = 0.f;
#pragma unroll
  for (int j = 0; j < 16; ++j) {
    v[j] = p[j];
    s   += v[j];
    s2   = fmaf(v[j], v[j], s2);
  }
  s  += __shfl_xor(s, 16, 32);
  s2 += __shfl_xor(s2, 16, 32);
  const float mean = s * (1.f / 32.f);
  const float var  = s2 * (1.f / 32.f) - mean * mean;
  const float inv  = rsqrtf(var + LN_EPS);
#pragma unroll
  for (int j = 0; j < 16; ++j) {
    const float t = (v[j] - mean) * inv * gLds[jb + j] + bLds[jb + j];
    p[j] = fmaxf(t, NEG_SLOPE * t);          // leaky-relu (slope > 0)
  }
}

// ---- build one B-fragment lane slice (16 f16) into the LDS weight pool ----
// wv fragments are prescaled by gamma: t = h@(g*wv) + (g*bv + h)
__device__ __forceinline__ void prep_frag(int t, _Float16* WF,
    const float* __restrict__ w1, const float* __restrict__ wv1,
    const float* __restrict__ w2, const float* __restrict__ wv2,
    const float* __restrict__ wo, float gamma1, float gamma2) {
  const int frag = t >> 5, lane = t & 31;
  const float* W; int ldn, kb, nb; float sc = 1.f;
  switch (frag) {
    case 0:  W = w1;  ldn = 32; kb = 0;  nb = 0;  break;
    case 1:  W = w1;  ldn = 32; kb = 32; nb = 0;  break;
    case 2:  W = w1;  ldn = 32; kb = 0;  nb = 16; break;
    case 3:  W = w1;  ldn = 32; kb = 32; nb = 16; break;
    case 4:  W = wv1; ldn = 32; kb = 0;  nb = 0;  sc = gamma1; break;
    case 5:  W = wv1; ldn = 32; kb = 0;  nb = 16; sc = gamma1; break;
    case 6:  W = w2;  ldn = 32; kb = 0;  nb = 0;  break;
    case 7:  W = w2;  ldn = 32; kb = 0;  nb = 16; break;
    case 8:  W = wv2; ldn = 32; kb = 0;  nb = 0;  sc = gamma2; break;
    case 9:  W = wv2; ldn = 32; kb = 0;  nb = 16; sc = gamma2; break;
    case 10: W = wo;  ldn = 64; kb = 0;  nb = 0;  break;
    case 11: W = wo;  ldn = 64; kb = 0;  nb = 16; break;
    case 12: W = wo;  ldn = 64; kb = 0;  nb = 32; break;
    default: W = wo;  ldn = 64; kb = 0;  nb = 48; break;
  }
  // B layout (32x16 f16): lanes 0-15 -> N=nb+lane, K=kb+{0..7,16..23}; lanes 16-31 -> K=kb+{8..15,24..31}
  const int n  = nb + (lane & 15);
  const int k0 = kb + ((lane >> 4) << 3);
  const float* p = W + (size_t)k0 * ldn + n;
  uint32_t* dst = reinterpret_cast<uint32_t*>(WF + frag * 512 + lane * 16);
#pragma unroll
  for (int e = 0; e < 16; e += 2) {
    union { _Float16 h[2]; uint32_t u; } pk;
    pk.h[0] = (_Float16)(sc * p[(size_t)(e + (e & 8)) * ldn]);
    pk.h[1] = (_Float16)(sc * p[(size_t)(e + 1 + (e & 8)) * ldn]);
    dst[e >> 1] = pk.u;
  }
}

__global__ __launch_bounds__(256)
void mlp_attn_wmma_kernel(
    const float* __restrict__ x,
    const float* __restrict__ w1,  const float* __restrict__ b1,
    const float* __restrict__ wv1, const float* __restrict__ bv1,
    const float* __restrict__ g1,
    const float* __restrict__ ln1g, const float* __restrict__ ln1b,
    const float* __restrict__ w2,  const float* __restrict__ b2,
    const float* __restrict__ wv2, const float* __restrict__ bv2,
    const float* __restrict__ g2,
    const float* __restrict__ ln2g, const float* __restrict__ ln2b,
    const float* __restrict__ wo,  const float* __restrict__ bo,
    float* __restrict__ out, int nrows)
{
  __shared__ __align__(32) _Float16 WF[NFRAG * 512];  // 14 KB: f16 B-fragments
  __shared__ float LNP[4][32];                        // ln1g, ln1b, ln2g, ln2b
  __shared__ float S[8][16 * LDSTRIDE];               // per-wave transpose tiles

  const int lane = threadIdx.x & 31;
  const int wave = threadIdx.x >> 5;
  const float gamma1 = g1[0];
  const float gamma2 = g2[0];

  // ---- block-cooperative prep: weights -> f16 WMMA fragments (gamma folded) ----
  for (int t = threadIdx.x; t < NFRAG * 32; t += 256)
    prep_frag(t, WF, w1, wv1, w2, wv2, wo, gamma1, gamma2);
  {
    const int t = threadIdx.x;
    if (t < 32)       LNP[0][t]      = ln1g[t];
    else if (t < 64)  LNP[1][t - 32] = ln1b[t - 32];
    else if (t < 96)  LNP[2][t - 64] = ln2g[t - 64];
    else if (t < 128) LNP[3][t - 96] = ln2b[t - 96];
  }
  __syncthreads();   // the ONLY workgroup barrier: WF/LNP are block-shared

  int r0 = (blockIdx.x * 8 + wave) * 16;
  const int rmax = nrows - 16;              // clamp keeps control flow uniform,
  r0 = (r0 > rmax) ? rmax : r0;             // EXEC all-ones for WMMA

  float* T = S[wave];                       // wave-private tile -> wave_sync only

  // ---------------- layer1: h1 = x @ w1 + b1   (K=64, N=32) ----------------
  const float* Xrow = x + (size_t)r0 * 64;
  const v16h ax0 = load_afrag(Xrow, 64, 0,  lane);
  const v16h ax1 = load_afrag(Xrow, 64, 32, lane);

  v8f c0 = bias_c(b1, 0, lane);
  c0 = wmma_f16(ax0, bfrag(WF, 0, lane), c0);
  c0 = wmma_f16(ax1, bfrag(WF, 1, lane), c0);
  v8f c1 = bias_c(b1, 16, lane);
  c1 = wmma_f16(ax0, bfrag(WF, 2, lane), c1);
  c1 = wmma_f16(ax1, bfrag(WF, 3, lane), c1);

  // transpose h1 through LDS -> A layout
  store_c(T, 0,  lane, c0);
  store_c(T, 16, lane, c1);
  wave_sync();
  const v16h ah1 = load_afrag(T, LDSTRIDE, 0, lane);
  wave_sync();

  // attn1 (T=1 => softmax==1): t = h1 @ (g1*wv1) + (g1*bv1 + h1)
  c0 = wmma_f16(ah1, bfrag(WF, 4, lane), resid_c(c0, gamma1, bv1, 0,  lane));
  c1 = wmma_f16(ah1, bfrag(WF, 5, lane), resid_c(c1, gamma1, bv1, 16, lane));

  store_c(T, 0,  lane, c0);
  store_c(T, 16, lane, c1);
  wave_sync();
  ln_leaky(T, LNP[0], LNP[1], lane);
  wave_sync();

  // ---------------- layer2: h2 = a1 @ w2 + b2  (K=32, N=32) ----------------
  const v16h aa1 = load_afrag(T, LDSTRIDE, 0, lane);
  v8f d0 = wmma_f16(aa1, bfrag(WF, 6, lane), bias_c(b2, 0,  lane));
  v8f d1 = wmma_f16(aa1, bfrag(WF, 7, lane), bias_c(b2, 16, lane));

  wave_sync();
  store_c(T, 0,  lane, d0);
  store_c(T, 16, lane, d1);
  wave_sync();
  const v16h ah2 = load_afrag(T, LDSTRIDE, 0, lane);

  // attn2: t2 = h2 @ (g2*wv2) + (g2*bv2 + h2)
  d0 = wmma_f16(ah2, bfrag(WF, 8, lane), resid_c(d0, gamma2, bv2, 0,  lane));
  d1 = wmma_f16(ah2, bfrag(WF, 9, lane), resid_c(d1, gamma2, bv2, 16, lane));

  wave_sync();
  store_c(T, 0,  lane, d0);
  store_c(T, 16, lane, d1);
  wave_sync();
  ln_leaky(T, LNP[2], LNP[3], lane);
  wave_sync();

  // ---------------- output: out = a2 @ wo + bo  (K=32, N=64) ----------------
  const v16h ao = load_afrag(T, LDSTRIDE, 0, lane);
  float* Orow = out + (size_t)r0 * 64;
  const int col = lane & 15;
  const int rb  = (lane >> 4) << 3;
#pragma unroll
  for (int q = 0; q < 4; ++q) {
    const v8f o = wmma_f16(ao, bfrag(WF, 10 + q, lane), bias_c(bo, q * 16, lane));
#pragma unroll
    for (int r = 0; r < 8; ++r)
      Orow[(size_t)(rb + r) * 64 + q * 16 + col] = o[r];
  }
}

extern "C" void kernel_launch(void* const* d_in, const int* in_sizes, int n_in,
                              void* d_out, int out_size, void* d_ws, size_t ws_size,
                              hipStream_t stream) {
  // setup_inputs() order:
  // 0:x 1:w1 2:b1 3:wq1 4:bq1 5:wk1 6:bk1 7:wv1 8:bv1 9:g1 10:ln1_g 11:ln1_b
  // 12:w2 13:b2 14:wq2 15:bq2 16:wk2 17:bk2 18:wv2 19:bv2 20:g2 21:ln2_g 22:ln2_b
  // 23:wo 24:bo      (wq/bq/wk/bk are dead for T=1: softmax of one element == 1)
  const float* x    = (const float*)d_in[0];
  const float* w1   = (const float*)d_in[1];
  const float* b1   = (const float*)d_in[2];
  const float* wv1  = (const float*)d_in[7];
  const float* bv1  = (const float*)d_in[8];
  const float* g1   = (const float*)d_in[9];
  const float* ln1g = (const float*)d_in[10];
  const float* ln1b = (const float*)d_in[11];
  const float* w2   = (const float*)d_in[12];
  const float* b2   = (const float*)d_in[13];
  const float* wv2  = (const float*)d_in[18];
  const float* bv2  = (const float*)d_in[19];
  const float* g2   = (const float*)d_in[20];
  const float* ln2g = (const float*)d_in[21];
  const float* ln2b = (const float*)d_in[22];
  const float* wo   = (const float*)d_in[23];
  const float* bo   = (const float*)d_in[24];
  float* out = (float*)d_out;

  const int nrows = in_sizes[0] / 64;          // B = 1,048,576
  const int blocks = (nrows + 127) / 128;      // 8 waves * 16 rows per block

  hipLaunchKernelGGL(mlp_attn_wmma_kernel, dim3(blocks), dim3(256), 0, stream,
                     x, w1, b1, wv1, bv1, g1, ln1g, ln1b,
                     w2, b2, wv2, bv2, g2, ln2g, ln2b, wo, bo, out, nrows);
}
